// Attention_25340307046737
// MI455X (gfx1250) — matrix operand order repaired and checked by reading, not compiled, not run on hardware
//
#include <hip/hip_runtime.h>
#include <hip/hip_bf16.h>
#include <math.h>

// Problem constants (match reference)
#define BB 2
#define SS 2048
#define DD 1024
#define HH 16
#define HD 64

typedef __attribute__((ext_vector_type(16))) _Float16 v16h;
typedef __attribute__((ext_vector_type(8)))  _Float16 v8h;
typedef __attribute__((ext_vector_type(8)))  float    v8f;
typedef __attribute__((ext_vector_type(4)))  float    v4f;

// ---------------------------------------------------------------------------
// WMMA helper: D = A(16x32 f16) x B(32x16 f16) + C(16x16 f32)
// ---------------------------------------------------------------------------
__device__ __forceinline__ v8f wmma_f32_f16(v16h a, v16h b, v8f c) {
  return __builtin_amdgcn_wmma_f32_16x16x32_f16(
      /*neg_a=*/false, a, /*neg_b=*/false, b,
      /*c_mod=*/(short)0, c, /*reuse_a=*/false, /*reuse_b=*/false);
}

// A-matrix (16x32 f16) lane loads per ISA layout:
//   lanes 0-15 : row M=lane,    K = kk+0..7   and kk+16..23
//   lanes 16-31: row M=lane-16, K = kk+8..15  and kk+24..31
__device__ __forceinline__ v16h load_a_f16(const _Float16* __restrict__ row,
                                           int kk, bool hi) {
  const int k0 = kk + (hi ? 8 : 0);
  v8h lo = *reinterpret_cast<const v8h*>(row + k0);
  v8h hp = *reinterpret_cast<const v8h*>(row + k0 + 16);
  v16h r;
#pragma unroll
  for (int i = 0; i < 8; ++i) { r[i] = lo[i]; r[i + 8] = hp[i]; }
  return r;
}

// Same A layout but source is fp32 (softmax probs) -> convert to f16 on load.
__device__ __forceinline__ v16h load_a_f32(const float* __restrict__ row,
                                           int kk, bool hi) {
  const int k0 = kk + (hi ? 8 : 0);
  v4f x0 = *reinterpret_cast<const v4f*>(row + k0);
  v4f x1 = *reinterpret_cast<const v4f*>(row + k0 + 4);
  v4f x2 = *reinterpret_cast<const v4f*>(row + k0 + 16);
  v4f x3 = *reinterpret_cast<const v4f*>(row + k0 + 20);
  v16h r;
#pragma unroll
  for (int i = 0; i < 4; ++i) {
    r[i]      = (_Float16)x0[i];
    r[i + 4]  = (_Float16)x1[i];
    r[i + 8]  = (_Float16)x2[i];
    r[i + 12] = (_Float16)x3[i];
  }
  return r;
}

// B-matrix (32x16 f16) lane loads per ISA layout:
//   lanes 0-15 : col N=lane,    K = kk+0..7   and kk+16..23
//   lanes 16-31: col N=lane-16, K = kk+8..15  and kk+24..31
__device__ __forceinline__ v16h load_b_f16(const _Float16* __restrict__ col,
                                           int kk, bool hi) {
  const int k0 = kk + (hi ? 8 : 0);
  v8h lo = *reinterpret_cast<const v8h*>(col + k0);
  v8h hp = *reinterpret_cast<const v8h*>(col + k0 + 16);
  v16h r;
#pragma unroll
  for (int i = 0; i < 8; ++i) { r[i] = lo[i]; r[i + 8] = hp[i]; }
  return r;
}

// ---------------------------------------------------------------------------
// fp32 -> f16 conversion (x and the four weight matrices)
// ---------------------------------------------------------------------------
__global__ void cvt_f32_to_f16(_Float16* __restrict__ dst,
                               const float* __restrict__ src, int n) {
  int i = blockIdx.x * blockDim.x + threadIdx.x;
  const int stride = gridDim.x * blockDim.x;
  for (; i < n; i += stride) dst[i] = (_Float16)src[i];
}

// ---------------------------------------------------------------------------
// QKV projection: [4096 x 1024] = x_h[4096 x 1024] * W^T (W is (N,K) row-major)
// blockIdx.z: 0 = Q (RoPE + 1/sqrt(HD) folded in), 1 = K (RoPE), 2 = V (store
// transposed as [b, h, hd, s] so the AV GEMM's B operand is K-contiguous).
// Block = 8 waves; wave computes a 16(M) x 64(N) tile (4 accumulators).
// ---------------------------------------------------------------------------
__global__ void qkv_gemm(const _Float16* __restrict__ x_h,
                         const _Float16* __restrict__ wq_h,
                         const _Float16* __restrict__ wk_h,
                         const _Float16* __restrict__ wv_h,
                         const float* __restrict__ cosp,
                         const float* __restrict__ sinp,
                         _Float16* __restrict__ q_h,
                         _Float16* __restrict__ k_h,
                         _Float16* __restrict__ v_t) {
  const int lane = threadIdx.x & 31;
  const int wave = threadIdx.x >> 5;
  const bool hi  = lane >= 16;
  const int l    = lane & 15;
  const int mode = blockIdx.z;
  const int m_tile = blockIdx.y * 128 + wave * 16;
  const int n_tile = blockIdx.x * 64;

  const _Float16* w = (mode == 0) ? wq_h : (mode == 1) ? wk_h : wv_h;
  const _Float16* arow = x_h + (size_t)(m_tile + l) * DD;

  v8f acc[4] = {};
  for (int kk = 0; kk < DD; kk += 32) {
    // Prefetch the A stream one iteration ahead (emits global_prefetch).
    if (kk + 32 < DD) __builtin_prefetch(arow + kk + 32, 0, 1);
    v16h a = load_a_f16(arow, kk, hi);
#pragma unroll
    for (int nt = 0; nt < 4; ++nt) {
      const int n = n_tile + nt * 16 + l;
      v16h bv = load_b_f16(w + (size_t)n * DD, kk, hi);
      acc[nt] = wmma_f32_f16(a, bv, acc[nt]);
    }
  }

  if (mode == 2) {
    // V: store transposed. C layout: lane owns column e, rows r -> s0..s0+7.
    const int bidx = m_tile >> 11;                 // / S
    const int s0   = (m_tile & (SS - 1)) + (hi ? 8 : 0);
#pragma unroll
    for (int nt = 0; nt < 4; ++nt) {
      const int e = n_tile + nt * 16 + l;          // e = h*64 + hd
      v8h pk;
#pragma unroll
      for (int r = 0; r < 8; ++r) pk[r] = (_Float16)acc[nt][r];
      *reinterpret_cast<v8h*>(v_t + ((size_t)(bidx * DD + e)) * SS + s0) = pk;
    }
  } else {
    _Float16* dst = (mode == 0) ? q_h : k_h;
    const float qscale = (mode == 0) ? 0.125f : 1.0f;  // 1/sqrt(64) folded in Q
#pragma unroll
    for (int nt = 0; nt < 4; ++nt) {
      const int e   = n_tile + nt * 16 + l;
      const int fi  = (e & 63) >> 1;               // RoPE frequency index
      const float sgn = (e & 1) ? 1.0f : -1.0f;    // even: r*c - i*s ; odd: r*s + i*c
#pragma unroll
      for (int r = 0; r < 8; ++r) {
        float val = acc[nt][r];
        float pv  = __shfl_xor(val, 1, 32);        // partner column e^1 = lane^1
        const int m = m_tile + r + (hi ? 8 : 0);
        const int s = m & (SS - 1);
        const float c  = cosp[s * (HD / 2) + fi];
        const float sn = sinp[s * (HD / 2) + fi];
        float o = (val * c + sgn * pv * sn) * qscale;
        dst[(size_t)m * DD + e] = (_Float16)o;
      }
    }
  }
}

// ---------------------------------------------------------------------------
// scores[b,h,q,k] = sum_hd q[b,q,h,hd] * k[b,k,h,hd]   (scale already in q)
// Per (b,h): M=2048, N=2048, K=64. Wave computes 16 x 128 (8 tiles, A reused).
// fp32 logits written straight into the attn_weights region of d_out.
// ---------------------------------------------------------------------------
__global__ void scores_gemm(const _Float16* __restrict__ q_h,
                            const _Float16* __restrict__ k_h,
                            float* __restrict__ attn) {
  const int lane = threadIdx.x & 31;
  const int wave = threadIdx.x >> 5;
  const bool hi  = lane >= 16;
  const int l    = lane & 15;
  const int bh   = blockIdx.z;
  const int b = bh >> 4, h = bh & 15;
  const int m_tile = blockIdx.y * 128 + wave * 16;
  const int n_tile = blockIdx.x * 128;

  const _Float16* arow = q_h + (size_t)(b * SS + m_tile + l) * DD + h * HD;
  v8f acc[8] = {};
#pragma unroll
  for (int kk = 0; kk < HD; kk += 32) {
    v16h a = load_a_f16(arow, kk, hi);
#pragma unroll
    for (int nt = 0; nt < 8; ++nt) {
      const int n = n_tile + nt * 16 + l;
      v16h bv = load_b_f16(k_h + (size_t)(b * SS + n) * DD + h * HD, kk, hi);
      acc[nt] = wmma_f32_f16(a, bv, acc[nt]);
    }
  }
#pragma unroll
  for (int nt = 0; nt < 8; ++nt) {
    const int n = n_tile + nt * 16 + l;
#pragma unroll
    for (int r = 0; r < 8; ++r) {
      const int m = m_tile + r + (hi ? 8 : 0);
      attn[((size_t)bh * SS + m) * SS + n] = acc[nt][r];
    }
  }
}

// ---------------------------------------------------------------------------
// Row softmax, in place. One block (256 threads) per row of 2048 fp32.
// ---------------------------------------------------------------------------
__global__ void softmax_rows(float* __restrict__ attn) {
  float* p = attn + (size_t)blockIdx.x * SS;
  const int t = threadIdx.x;
  __shared__ float sh[8];

  float v[8];
  float m = -3.402823466e38f;
#pragma unroll
  for (int i = 0; i < 8; ++i) { v[i] = p[t + 256 * i]; m = fmaxf(m, v[i]); }
#pragma unroll
  for (int off = 16; off > 0; off >>= 1) m = fmaxf(m, __shfl_xor(m, off, 32));
  if ((t & 31) == 0) sh[t >> 5] = m;
  __syncthreads();
  float bm = sh[0];
#pragma unroll
  for (int i = 1; i < 8; ++i) bm = fmaxf(bm, sh[i]);
  __syncthreads();

  float s = 0.0f;
#pragma unroll
  for (int i = 0; i < 8; ++i) { v[i] = expf(v[i] - bm); s += v[i]; }
#pragma unroll
  for (int off = 16; off > 0; off >>= 1) s += __shfl_xor(s, off, 32);
  if ((t & 31) == 0) sh[t >> 5] = s;
  __syncthreads();
  float tot = 0.0f;
#pragma unroll
  for (int i = 0; i < 8; ++i) tot += sh[i];
  const float inv = 1.0f / tot;
#pragma unroll
  for (int i = 0; i < 8; ++i) p[t + 256 * i] = v[i] * inv;
}

// ---------------------------------------------------------------------------
// out[b,q,h,hd] = sum_k probs[b,h,q,k] * v[b,k,h,hd]
// Per (b,h): M=2048, N=64, K=2048. A = probs fp32 -> f16 on load; this is the
// one true HBM-streaming read (512 MB, read exactly once) -> prefetch ahead.
// B = v_t[b,h,hd,s] (K-contiguous). Wave computes 16 x 64 (4 tiles).
// ---------------------------------------------------------------------------
__global__ void av_gemm(const float* __restrict__ attn,
                        const _Float16* __restrict__ v_t,
                        _Float16* __restrict__ ao_h) {
  const int lane = threadIdx.x & 31;
  const int wave = threadIdx.x >> 5;
  const bool hi  = lane >= 16;
  const int l    = lane & 15;
  const int bh   = blockIdx.y;
  const int b = bh >> 4, h = bh & 15;
  const int m_tile = blockIdx.x * 128 + wave * 16;

  const float* arow = attn + ((size_t)bh * SS + m_tile + l) * SS;
  const _Float16* vb = v_t + (size_t)(b * DD + h * HD) * SS;

  v8f acc[4] = {};
  for (int kk = 0; kk < SS; kk += 32) {
    // Prefetch the probs stream ~2 iterations ahead (emits global_prefetch).
    if (kk + 64 < SS) __builtin_prefetch(arow + kk + 64, 0, 1);
    v16h a = load_a_f32(arow, kk, hi);
#pragma unroll
    for (int nt = 0; nt < 4; ++nt) {
      const int n = nt * 16 + l;                   // hd
      v16h bv = load_b_f16(vb + (size_t)n * SS, kk, hi);
      acc[nt] = wmma_f32_f16(a, bv, acc[nt]);
    }
  }
#pragma unroll
  for (int nt = 0; nt < 4; ++nt) {
    const int e = h * HD + nt * 16 + l;
#pragma unroll
    for (int r = 0; r < 8; ++r) {
      const int m = m_tile + r + (hi ? 8 : 0);
      ao_h[(size_t)(b * SS + m) * DD + e] = (_Float16)acc[nt][r];
    }
  }
}

// ---------------------------------------------------------------------------
// Output projection: out[m, d] = sum_e ao[m, e] * wo[d, e]  (fp32 result)
// ---------------------------------------------------------------------------
__global__ void proj_gemm(const _Float16* __restrict__ ao_h,
                          const _Float16* __restrict__ wo_h,
                          float* __restrict__ out) {
  const int lane = threadIdx.x & 31;
  const int wave = threadIdx.x >> 5;
  const bool hi  = lane >= 16;
  const int l    = lane & 15;
  const int m_tile = blockIdx.y * 128 + wave * 16;
  const int n_tile = blockIdx.x * 64;

  const _Float16* arow = ao_h + (size_t)(m_tile + l) * DD;
  v8f acc[4] = {};
  for (int kk = 0; kk < DD; kk += 32) {
    if (kk + 32 < DD) __builtin_prefetch(arow + kk + 32, 0, 1);
    v16h a = load_a_f16(arow, kk, hi);
#pragma unroll
    for (int nt = 0; nt < 4; ++nt) {
      const int n = n_tile + nt * 16 + l;
      v16h bv = load_b_f16(wo_h + (size_t)n * DD, kk, hi);
      acc[nt] = wmma_f32_f16(a, bv, acc[nt]);
    }
  }
#pragma unroll
  for (int nt = 0; nt < 4; ++nt) {
    const int n = n_tile + nt * 16 + l;
#pragma unroll
    for (int r = 0; r < 8; ++r) {
      const int m = m_tile + r + (hi ? 8 : 0);
      out[(size_t)m * DD + n] = acc[nt][r];
    }
  }
}

// ---------------------------------------------------------------------------
// Launch. Inputs: x, cos, sin, wq, wk, wv, wo (all fp32).
// d_out = out (2*2048*1024 fp32) ++ attn_weights (2*16*2048*2048 fp32).
// Workspace layout (48 MB total): wq|wk|wv|wo f16 (2 MB each), x_h (8 MB),
// q_h (8 MB), k_h (8 MB), v_t (8 MB), ao_h (8 MB).
// ---------------------------------------------------------------------------
extern "C" void kernel_launch(void* const* d_in, const int* in_sizes, int n_in,
                              void* d_out, int out_size, void* d_ws, size_t ws_size,
                              hipStream_t stream) {
  const float* x    = (const float*)d_in[0];
  const float* cosp = (const float*)d_in[1];
  const float* sinp = (const float*)d_in[2];
  const float* wq   = (const float*)d_in[3];
  const float* wk   = (const float*)d_in[4];
  const float* wv   = (const float*)d_in[5];
  const float* wo   = (const float*)d_in[6];

  float* out  = (float*)d_out;
  float* attn = out + (size_t)BB * SS * DD;        // +4,194,304

  char* ws = (char*)d_ws;
  const size_t MB = 1024 * 1024;
  _Float16* wq_h = (_Float16*)(ws + 0 * MB);
  _Float16* wk_h = (_Float16*)(ws + 2 * MB);
  _Float16* wv_h = (_Float16*)(ws + 4 * MB);
  _Float16* wo_h = (_Float16*)(ws + 6 * MB);
  _Float16* x_h  = (_Float16*)(ws + 8 * MB);
  _Float16* q_h  = (_Float16*)(ws + 16 * MB);
  _Float16* k_h  = (_Float16*)(ws + 24 * MB);
  _Float16* v_t  = (_Float16*)(ws + 32 * MB);
  _Float16* ao_h = (_Float16*)(ws + 40 * MB);

  // fp32 -> f16 conversions
  cvt_f32_to_f16<<<4096, 256, 0, stream>>>(x_h, x, BB * SS * DD);
  cvt_f32_to_f16<<<1024, 256, 0, stream>>>(wq_h, wq, DD * DD);
  cvt_f32_to_f16<<<1024, 256, 0, stream>>>(wk_h, wk, DD * DD);
  cvt_f32_to_f16<<<1024, 256, 0, stream>>>(wv_h, wv, DD * DD);
  cvt_f32_to_f16<<<1024, 256, 0, stream>>>(wo_h, wo, DD * DD);

  // Q/K/V projections (+RoPE, +scale on Q, V transposed)
  qkv_gemm<<<dim3(DD / 64, (BB * SS) / 128, 3), 256, 0, stream>>>(
      x_h, wq_h, wk_h, wv_h, cosp, sinp, q_h, k_h, v_t);

  // logits -> attn region of d_out
  scores_gemm<<<dim3(SS / 128, SS / 128, BB * HH), 256, 0, stream>>>(q_h, k_h, attn);

  // softmax in place (one block per row)
  softmax_rows<<<BB * HH * SS, 256, 0, stream>>>(attn);

  // probs x V
  av_gemm<<<dim3(SS / 128, BB * HH), 256, 0, stream>>>(attn, v_t, ao_h);

  // output projection
  proj_gemm<<<dim3(DD / 64, (BB * SS) / 128), 256, 0, stream>>>(ao_h, wo_h, out);
}
